// Ours_Net_17815524344175
// MI455X (gfx1250) — compile-verified, matrix operand order checked by reading, not run on hardware
//
#include <hip/hip_runtime.h>
#include <hip/hip_bf16.h>

// ---------------------------------------------------------------------------
// ViT-B (12 blocks) forward with prefix-tuning prompts, CDNA5 / gfx1250.
// All GEMMs + attention on v_wmma_f32_16x16x32_f16 (f16 in, f32 accumulate).
// Weights + inter-layer activations are kept as f16 in global memory; tile
// staging uses the CDNA5 async global->LDS path (ASYNCcnt) with LDS double
// buffering so DMA overlaps the WMMA stream. Residual stream stays fp32.
// ---------------------------------------------------------------------------

typedef _Float16 v16h __attribute__((ext_vector_type(16)));
typedef float    v8f  __attribute__((ext_vector_type(8)));
typedef int      v4i  __attribute__((vector_size(16)));  // b128 payload type

#if defined(__AMDGCN__) &&                                                   \
    __has_builtin(__builtin_amdgcn_global_load_async_to_lds_b128) &&         \
    __has_builtin(__builtin_amdgcn_s_wait_asynccnt)
#define USE_ASYNC_LDS 1
#else
#define USE_ASYNC_LDS 0
#endif

// Copy 16 bytes (8 halves) global -> LDS. Async (ASYNCcnt-tracked) when the
// toolchain exposes the gfx1250 builtins, else plain load + ds_store.
__device__ __forceinline__ void cp16_g2l(const _Float16* g, _Float16* l) {
#if USE_ASYNC_LDS
  __builtin_amdgcn_global_load_async_to_lds_b128(
      (__attribute__((address_space(1))) v4i*)g,
      (__attribute__((address_space(3))) v4i*)l, 0, 0);
#else
  *(uint4*)l = *(const uint4*)g;
#endif
}

__device__ __forceinline__ void zero16_l(_Float16* l) {
  *(uint4*)l = make_uint4(0u, 0u, 0u, 0u);
}

__device__ __forceinline__ void wait_async_lds() {
#if USE_ASYNC_LDS
  __builtin_amdgcn_s_wait_asynccnt(0);
#endif
}

__device__ __forceinline__ v8f wmma32(v16h a, v16h b, v8f c) {
  // 8 args: (neg_a, A, neg_b, B, c_mod, C, reuse_a, reuse_b)
  return __builtin_amdgcn_wmma_f32_16x16x32_f16(false, a, false, b, (short)0, c,
                                                false, false);
}

// Load a 16x32 f16 fragment (A-layout; B uses the same layout when the source
// is stored [n][k] row-major). ISA 7.12.2: lanes 0-15 hold row M=lane with
// K pairs {0,1},{2,3},{4,5},{6,7},{16,17},... ; lanes 16-31 hold K offset +8.
__device__ __forceinline__ v16h frag_ld(const _Float16* __restrict__ base,
                                        int row0, int ldh, int kbase) {
  const int lane = threadIdx.x & 31;
  const int hs = (lane >> 4) << 3;  // 0 or 8
  const unsigned* up =
      (const unsigned*)(base + (size_t)(row0 + (lane & 15)) * ldh + kbase);
  union { v16h v; unsigned u[8]; } f;
#pragma unroll
  for (int j = 0; j < 8; ++j) {
    const int k = ((j & 4) << 2) + hs + ((j & 3) << 1);
    f.u[j] = up[k >> 1];
  }
  return f.v;
}

__device__ __forceinline__ float gelu_exact(float x) {
  return 0.5f * x * (1.0f + erff(x * 0.70710678118654752f));
}

union U4H8 { uint4 u; _Float16 h[8]; };

// ---------------------------------------------------------------------------
// GEMM: out[M,N] = act(A[M,K] @ W[N,K]^T + bias[N] (+ res[M,N]))
// A, W are f16; out is f16 (outH) or f32 (outF). Block tile 128x128, BK=32,
// 256 threads = 8 waves, each wave a 64x32 subtile (8 WMMA accum tiles).
// LDS double-buffered: async copies for chunk k+1 fly during chunk k's WMMAs;
// one barrier per K-chunk.
// ---------------------------------------------------------------------------
__global__ __launch_bounds__(256) void gemm_wmma(
    const _Float16* __restrict__ A, const _Float16* __restrict__ W,
    const float* __restrict__ bias, const float* __restrict__ res,
    float* __restrict__ outF, _Float16* __restrict__ outH, int M, int N, int K,
    int act) {
  __shared__ _Float16 As[2][128 * 48];  // 96B rows, 16B-aligned
  __shared__ _Float16 Bs[2][128 * 48];

  const int t = threadIdx.x;
  const int wave = t >> 5;
  const int waveM = (wave >> 2) * 64;  // 0 or 64
  const int waveN = (wave & 3) * 32;   // 0..96

  const int srow = t >> 1;             // 0..127
  const int shalf = (t & 1) << 4;      // 0 or 16 (halves)
  const int gmA = blockIdx.y * 128 + srow;
  const int gnB = blockIdx.x * 128 + srow;
  const _Float16* ap = A + (size_t)gmA * K + shalf;
  const _Float16* bp = W + (size_t)gnB * K + shalf;

  auto stage = [&](int kc, int buf) {
    _Float16* la = As[buf] + srow * 48 + shalf;
    _Float16* lb = Bs[buf] + srow * 48 + shalf;
    if (gmA < M) {
      cp16_g2l(ap + kc, la);
      cp16_g2l(ap + kc + 8, la + 8);
      if (kc + 32 < K) __builtin_prefetch(ap + kc + 32, 0, 1);
    } else {
      zero16_l(la);
      zero16_l(la + 8);
    }
    if (gnB < N) {
      cp16_g2l(bp + kc, lb);
      cp16_g2l(bp + kc + 8, lb + 8);
      if (kc + 32 < K) __builtin_prefetch(bp + kc + 32, 0, 1);
    } else {
      zero16_l(lb);
      zero16_l(lb + 8);
    }
  };

  const v8f vzero = {0.f, 0.f, 0.f, 0.f, 0.f, 0.f, 0.f, 0.f};
  v8f acc[4][2];
#pragma unroll
  for (int mt = 0; mt < 4; ++mt)
#pragma unroll
    for (int nt = 0; nt < 2; ++nt) acc[mt][nt] = vzero;

  stage(0, 0);  // prologue: chunk 0 into buffer 0

  int buf = 0;
  for (int kc = 0; kc < K; kc += 32, buf ^= 1) {
    wait_async_lds();   // this wave's copies for `buf` have landed
    __syncthreads();    // all waves' copies landed; prev reads of buf^1 done
    if (kc + 32 < K) stage(kc + 32, buf ^ 1);  // DMA next chunk during WMMAs

    v16h afr[4], bfr[2];
#pragma unroll
    for (int mt = 0; mt < 4; ++mt)
      afr[mt] = frag_ld(As[buf], waveM + mt * 16, 48, 0);
#pragma unroll
    for (int nt = 0; nt < 2; ++nt)
      bfr[nt] = frag_ld(Bs[buf], waveN + nt * 16, 48, 0);
#pragma unroll
    for (int mt = 0; mt < 4; ++mt)
#pragma unroll
      for (int nt = 0; nt < 2; ++nt)
        acc[mt][nt] = wmma32(afr[mt], bfr[nt], acc[mt][nt]);
  }

  // C/D layout: VGPR r -> row r (lanes 0-15) or r+8 (lanes 16-31).
  const int lane = t & 31;
  const int nloc = lane & 15;
  const int moff = (lane >> 4) << 3;
#pragma unroll
  for (int mt = 0; mt < 4; ++mt)
#pragma unroll
    for (int nt = 0; nt < 2; ++nt) {
      const int gn = blockIdx.x * 128 + waveN + nt * 16 + nloc;
      const float bval = bias[gn];
#pragma unroll
      for (int r = 0; r < 8; ++r) {
        const int gm = blockIdx.y * 128 + waveM + mt * 16 + r + moff;
        if (gm < M) {
          float vv = acc[mt][nt][r] + bval;
          if (res) vv += res[(size_t)gm * N + gn];
          if (act) vv = gelu_exact(vv);
          if (outH) outH[(size_t)gm * N + gn] = (_Float16)vv;
          else      outF[(size_t)gm * N + gn] = vv;
        }
      }
    }
}

// ---------------------------------------------------------------------------
// Attention: one workgroup per (b, h, 32-query tile); q/k/v/o are f16 with
// layout [B, tokens, C], C = H*64, head h at channel h*64.
// ---------------------------------------------------------------------------
__global__ __launch_bounds__(256) void attn_wmma(
    const _Float16* __restrict__ q, const _Float16* __restrict__ k,
    const _Float16* __restrict__ v, _Float16* __restrict__ o, int Nq, int Nk,
    int C, int H, float scale) {
  __shared__ _Float16 Qs[32 * 72];   // [qrow][d]   (144B rows, 16B-aligned)
  __shared__ _Float16 Ks[64 * 72];   // [key][d]
  __shared__ _Float16 Vs[64 * 40];   // [d][key]    (transposed chunk)
  __shared__ _Float16 Sh[32 * 264];  // [qrow][key] scores -> probs

  const int t = threadIdx.x;
  const int wave = t >> 5;
  const int bh = blockIdx.x;
  const int b = bh / H, h = bh % H;
  const int q0 = blockIdx.y * 32;
  const int Nkc = (Nk + 63) >> 6;  // 64-key chunks
  const int Nkp = Nkc << 6;        // padded key count (<=256)
  const size_t qbase = (size_t)b * Nq * C + (size_t)h * 64;
  const size_t kvbase = (size_t)b * Nk * C + (size_t)h * 64;
  const v8f vzero = {0.f, 0.f, 0.f, 0.f, 0.f, 0.f, 0.f, 0.f};

  // Stage Q tile (32 x 64 halves) — async 16B chunks.
  {
    const int r = t >> 3;            // 0..31
    const int seg = (t & 7) << 3;    // 0..56 halves
    const _Float16* g = q + qbase + (size_t)(q0 + r) * C + seg;
    _Float16* l = Qs + r * 72 + seg;
    if (q0 + r < Nq) cp16_g2l(g, l);
    else             zero16_l(l);
  }

  const int mt = (wave & 1) << 4;   // query subtile: 0 or 16
  const int nt = (wave >> 1) << 4;  // col subtile:   0..48
  const int lane = t & 31;
  const int nloc = lane & 15;
  const int moff = (lane >> 4) << 3;

  // S = Q K^T over 64-key chunks; 8 waves cover each 32x64 score chunk.
  for (int kc = 0; kc < Nkc; ++kc) {
    {
      const int j = t >> 2;          // 0..63
      const int seg = (t & 3) << 4;  // 0,16,32,48 halves
      const int grow = (kc << 6) + j;
      const _Float16* g = k + kvbase + (size_t)grow * C + seg;
      _Float16* l = Ks + j * 72 + seg;
      if (grow < Nk) {
        cp16_g2l(g, l);
        cp16_g2l(g + 8, l + 8);
      } else {
        zero16_l(l);
        zero16_l(l + 8);
      }
    }
    wait_async_lds();
    __syncthreads();

    v16h a0 = frag_ld(Qs, mt, 72, 0);
    v16h a1 = frag_ld(Qs, mt, 72, 32);
    v16h b0 = frag_ld(Ks, nt, 72, 0);
    v16h b1 = frag_ld(Ks, nt, 72, 32);
    v8f s = vzero;
    s = wmma32(a0, b0, s);
    s = wmma32(a1, b1, s);
#pragma unroll
    for (int r = 0; r < 8; ++r) {
      const int m = mt + r + moff;
      const int col = (kc << 6) + nt + nloc;
      float val = s[r] * scale;
      if (col >= Nk) val = -60000.0f;  // mask padded keys
      Sh[m * 264 + col] = (_Float16)val;
    }
    __syncthreads();
  }

  // Row softmax (wave 0, one lane per query row, full EXEC).
  if (t < 32) {
    const int rb = t * 264;
    float mx = -1e30f;
    for (int c = 0; c < Nkp; ++c) mx = fmaxf(mx, (float)Sh[rb + c]);
    float sum = 0.f;
    for (int c = 0; c < Nkp; ++c) {
      const float e = expf((float)Sh[rb + c] - mx);
      sum += e;
      Sh[rb + c] = (_Float16)e;
    }
    const float inv = 1.0f / sum;
    for (int c = 0; c < Nkp; ++c)
      Sh[rb + c] = (_Float16)((float)Sh[rb + c] * inv);
  }
  __syncthreads();

  // O = P V over 32-key chunks; V staged transposed ([d][key]) so B-fragments
  // read the [n][k] layout.
  v8f oacc = vzero;
  for (int kc2 = 0; kc2 < Nkp; kc2 += 32) {
    {
      const int jj = t >> 3;         // key within chunk: 0..31
      const int d0 = (t & 7) << 3;   // 0..56
      const int grow = kc2 + jj;
      U4H8 pv;
      if (grow < Nk)
        pv.u = *(const uint4*)(v + kvbase + (size_t)grow * C + d0);
      else
        pv.u = make_uint4(0u, 0u, 0u, 0u);
#pragma unroll
      for (int i = 0; i < 8; ++i) Vs[(d0 + i) * 40 + jj] = pv.h[i];
    }
    __syncthreads();
    v16h a = frag_ld(Sh, mt, 264, kc2);
    v16h bf = frag_ld(Vs, nt, 40, 0);
    oacc = wmma32(a, bf, oacc);
    __syncthreads();
  }

#pragma unroll
  for (int r = 0; r < 8; ++r) {
    const int m = mt + r + moff;  // local query row (0..31)
    const int d = nt + nloc;      // head dim (0..63)
    if (q0 + m < Nq)
      o[qbase + (size_t)(q0 + m) * C + d] = (_Float16)oacc[r];
  }
}

// ---------------------------------------------------------------------------
// LayerNorm over fp32 rows; writes f16 (yH) or fp32 (yF).
// ---------------------------------------------------------------------------
__global__ __launch_bounds__(256) void ln_kernel(const float* __restrict__ x,
                                                 const float* __restrict__ w,
                                                 const float* __restrict__ b,
                                                 float* __restrict__ yF,
                                                 _Float16* __restrict__ yH,
                                                 int C) {
  __shared__ float red[256];
  __shared__ float stat[2];
  const int t = threadIdx.x;
  const size_t base = (size_t)blockIdx.x * C;

  float s = 0.f;
  for (int c = t; c < C; c += 256) s += x[base + c];
  red[t] = s;
  __syncthreads();
  for (int o2 = 128; o2 > 0; o2 >>= 1) {
    if (t < o2) red[t] += red[t + o2];
    __syncthreads();
  }
  if (t == 0) stat[0] = red[0] / (float)C;
  __syncthreads();
  const float mean = stat[0];

  float vs = 0.f;
  for (int c = t; c < C; c += 256) {
    const float d = x[base + c] - mean;
    vs += d * d;
  }
  red[t] = vs;
  __syncthreads();
  for (int o2 = 128; o2 > 0; o2 >>= 1) {
    if (t < o2) red[t] += red[t + o2];
    __syncthreads();
  }
  if (t == 0) stat[1] = rsqrtf(red[0] / (float)C + 1e-6f);
  __syncthreads();
  const float inv = stat[1];

  for (int c = t; c < C; c += 256) {
    const float r = (x[base + c] - mean) * inv * w[c] + b[c];
    if (yH) yH[base + c] = (_Float16)r;
    else    yF[base + c] = r;
  }
}

// ---------------------------------------------------------------------------
// Build prefix-augmented K/V input streams (f16): [B, Nq+PL, C].
// pp (fp32): this layer's prompt [2*PL, C] (rows 0..PL-1 = pk, rest = pv).
// ---------------------------------------------------------------------------
__global__ void build_kv(const _Float16* __restrict__ xn,
                         const float* __restrict__ pp,
                         _Float16* __restrict__ kin,
                         _Float16* __restrict__ vin, int Bn, int Nq, int PLen,
                         int C) {
  const int Ntot = Nq + PLen;
  const size_t idx = (size_t)blockIdx.x * blockDim.x + threadIdx.x;
  const size_t total = (size_t)Bn * Ntot * C;
  if (idx >= total) return;
  const int c = (int)(idx % C);
  const size_t rt = idx / C;
  const int tt = (int)(rt % Ntot);
  const int b = (int)(rt / Ntot);
  if (tt < Nq) {
    const _Float16 xv = xn[((size_t)b * Nq + tt) * C + c];
    kin[idx] = xv;
    vin[idx] = xv;
  } else {
    kin[idx] = (_Float16)pp[(size_t)(tt - Nq) * C + c];
    vin[idx] = (_Float16)pp[(size_t)(PLen + tt - Nq) * C + c];
  }
}

// fp32 -> f16 bulk conversion (weights).
__global__ void cvt_f32_f16(const float* __restrict__ s,
                            _Float16* __restrict__ d, size_t n) {
  const size_t i = (size_t)blockIdx.x * 256 + threadIdx.x;
  if (i < n) d[i] = (_Float16)s[i];
}

// ---------------------------------------------------------------------------
// Host orchestration.
// ---------------------------------------------------------------------------
extern "C" void kernel_launch(void* const* d_in, const int* in_sizes, int n_in,
                              void* d_out, int out_size, void* d_ws,
                              size_t ws_size, hipStream_t stream) {
  (void)in_sizes; (void)n_in; (void)out_size; (void)ws_size;

  const float* x_in   = (const float*)d_in[0];
  const float* qkv_w  = (const float*)d_in[1];
  const float* qkv_b  = (const float*)d_in[2];
  const float* proj_w = (const float*)d_in[3];
  const float* proj_b = (const float*)d_in[4];
  const float* n1_w   = (const float*)d_in[5];
  const float* n1_b   = (const float*)d_in[6];
  const float* n2_w   = (const float*)d_in[7];
  const float* n2_b   = (const float*)d_in[8];
  const float* fc1_w  = (const float*)d_in[9];
  const float* fc1_b  = (const float*)d_in[10];
  const float* fc2_w  = (const float*)d_in[11];
  const float* fc2_b  = (const float*)d_in[12];
  const float* prompt = (const float*)d_in[13];
  const float* nf_w   = (const float*)d_in[14];
  const float* nf_b   = (const float*)d_in[15];

  const int Bn = 64, N = 201, C = 768, H = 12, L = 12, PL = 5;
  const int Nk2 = N + PL;               // 206
  const size_t MN = (size_t)Bn * N;     // 12864
  const size_t MN2 = (size_t)Bn * Nk2;  // 13184
  const float scale = 0.125f;           // 64^-0.5

  char* wsb = (char*)d_ws;
  size_t off = 0;
  auto alloc = [&](size_t bytes) -> void* {
    void* p = wsb + off;
    off += (bytes + 255) & ~(size_t)255;
    return p;
  };
  float*    xbuf = (float*)alloc(MN * C * 4);
  _Float16* xn   = (_Float16*)alloc(MN * C * 2);
  _Float16* kin  = (_Float16*)alloc(MN2 * C * 2);
  _Float16* vin  = (_Float16*)alloc(MN2 * C * 2);
  _Float16* qb   = (_Float16*)alloc(MN * C * 2);
  _Float16* kb   = (_Float16*)alloc(MN2 * C * 2);
  _Float16* vb   = (_Float16*)alloc(MN2 * C * 2);
  _Float16* ao   = (_Float16*)alloc(MN * C * 2);
  _Float16* hb   = (_Float16*)alloc(MN * 4 * C * 2);
  const size_t nQKV = (size_t)L * 3 * C * C;
  const size_t nPRJ = (size_t)L * C * C;
  const size_t nFC  = (size_t)L * 4 * C * C;
  _Float16* qkvWh = (_Float16*)alloc(nQKV * 2);
  _Float16* prjWh = (_Float16*)alloc(nPRJ * 2);
  _Float16* fc1Wh = (_Float16*)alloc(nFC * 2);
  _Float16* fc2Wh = (_Float16*)alloc(nFC * 2);

  (void)hipMemcpyAsync(xbuf, x_in, MN * C * sizeof(float),
                       hipMemcpyDeviceToDevice, stream);

  const dim3 blk(256);
  cvt_f32_f16<<<(int)((nQKV + 255) / 256), blk, 0, stream>>>(qkv_w, qkvWh, nQKV);
  cvt_f32_f16<<<(int)((nPRJ + 255) / 256), blk, 0, stream>>>(proj_w, prjWh, nPRJ);
  cvt_f32_f16<<<(int)((nFC + 255) / 256), blk, 0, stream>>>(fc1_w, fc1Wh, nFC);
  cvt_f32_f16<<<(int)((nFC + 255) / 256), blk, 0, stream>>>(fc2_w, fc2Wh, nFC);

  const int mBlocks  = (int)((MN + 127) / 128);
  const int mBlocks2 = (int)((MN2 + 127) / 128);

  for (int l = 0; l < L; ++l) {
    ln_kernel<<<(int)MN, blk, 0, stream>>>(xbuf, n1_w + (size_t)l * C,
                                           n1_b + (size_t)l * C, nullptr, xn, C);

    const _Float16* Wqkv = qkvWh + (size_t)l * 3 * C * C;
    const float*    bqkv = qkv_b + (size_t)l * 3 * C;

    const _Float16* kinp = xn;
    const _Float16* vinp = xn;
    int Nk = N;
    int mkB = mBlocks;
    if (l < PL) {  // PRE_POS = layers 0..4
      const size_t total = (size_t)Bn * Nk2 * C;
      build_kv<<<(int)((total + 255) / 256), blk, 0, stream>>>(
          xn, prompt + (size_t)l * 2 * PL * C, kin, vin, Bn, N, PL, C);
      kinp = kin; vinp = vin; Nk = Nk2; mkB = mBlocks2;
    }
    const int Mkv = Bn * Nk;

    gemm_wmma<<<dim3(C / 128, mBlocks), blk, 0, stream>>>(
        xn, Wqkv, bqkv, nullptr, nullptr, qb, (int)MN, C, C, 0);
    gemm_wmma<<<dim3(C / 128, mkB), blk, 0, stream>>>(
        kinp, Wqkv + (size_t)C * C, bqkv + C, nullptr, nullptr, kb, Mkv, C, C, 0);
    gemm_wmma<<<dim3(C / 128, mkB), blk, 0, stream>>>(
        vinp, Wqkv + (size_t)2 * C * C, bqkv + 2 * C, nullptr, nullptr, vb, Mkv,
        C, C, 0);

    attn_wmma<<<dim3(Bn * H, (N + 31) / 32), blk, 0, stream>>>(
        qb, kb, vb, ao, N, Nk, C, H, scale);

    // proj + residual (fp32, in-place into xbuf).
    gemm_wmma<<<dim3(C / 128, mBlocks), blk, 0, stream>>>(
        ao, prjWh + (size_t)l * C * C, proj_b + (size_t)l * C, xbuf, xbuf,
        nullptr, (int)MN, C, C, 0);

    ln_kernel<<<(int)MN, blk, 0, stream>>>(xbuf, n2_w + (size_t)l * C,
                                           n2_b + (size_t)l * C, nullptr, xn, C);

    gemm_wmma<<<dim3(4 * C / 128, mBlocks), blk, 0, stream>>>(
        xn, fc1Wh + (size_t)l * 4 * C * C, fc1_b + (size_t)l * 4 * C, nullptr,
        nullptr, hb, (int)MN, 4 * C, C, /*gelu=*/1);

    gemm_wmma<<<dim3(C / 128, mBlocks), blk, 0, stream>>>(
        hb, fc2Wh + (size_t)l * C * 4 * C, fc2_b + (size_t)l * C, xbuf, xbuf,
        nullptr, (int)MN, C, 4 * C, 0);
  }

  ln_kernel<<<(int)MN, blk, 0, stream>>>(xbuf, nf_w, nf_b, (float*)d_out,
                                         nullptr, C);
}
